// hist_hTCN_22162031247535
// MI455X (gfx1250) — matrix operand order repaired and checked by reading, not compiled, not run on hardware
//
#include <hip/hip_runtime.h>
#include <hip/hip_bf16.h>

typedef __attribute__((ext_vector_type(16))) _Float16 v16h;
typedef __attribute__((ext_vector_type(4)))  _Float16 v4h;
typedef __attribute__((ext_vector_type(8)))  float    v8f;
typedef __attribute__((ext_vector_type(4)))  unsigned int v4u;
typedef __attribute__((ext_vector_type(8)))  int      v8i;
typedef __attribute__((ext_vector_type(4)))  int      v4i;

#define SUB_NO 20
#define M_NO   10
#define B_NO   12
#define T_NO   200
#define C_NO   (SUB_NO * M_NO)   // 200 channels

// Use the raw 5-arg TDM builtin only on the ROCm-7.2-style toolchain (the
// therock toolchain ships the gfx1250 TDM header and a 6-arg builtin).
#if !defined(__HIP_DEVICE_COMPILE__) || 1
#if !__has_include(<hip/amd_detail/amd_gfx1250_TDM.h>) && \
    __has_builtin(__builtin_amdgcn_tensor_load_to_lds) && \
    __has_builtin(__builtin_amdgcn_s_wait_tensorcnt)
#define USE_TDM 1
#endif
#endif

// ---------------------------------------------------------------------------
// Kernel 1: WMMA GEMM  out[s,t] = sum_e C[s,e] * S[t,e]
//   M = 20 (padded to 32), N = T_data, K = Kdim.  f32 -> f16 conversion fused
//   into fragment-ordered LDS staging; f32 accumulate via v_wmma_f32_16x16x32_f16.
//   Block: 256 threads = 8 waves; wave w -> (mtile = w&1, ntile = w>>1),
//   covering a 32 x 64 output tile per block.
//   Fast path (block-uniform): float4 global loads -> packed v4h ds_store_b64.
//   A-pad rows (m >= 20) are zeroed once before the k-loop (no divergent guard).
// ---------------------------------------------------------------------------
__global__ void syn_gemm_wmma(const float* __restrict__ S,   // (T, Kdim)
                              const float* __restrict__ C,   // (20, Kdim)
                              float* __restrict__ out,       // (20, T)
                              int Kdim, int T) {
  __shared__ __align__(32) _Float16 afrag[2][512];
  __shared__ __align__(32) _Float16 bfrag[4][512];

  const int tid   = threadIdx.x;
  const int lane  = tid & 31;
  const int wid   = tid >> 5;
  const int mtile = wid & 1;
  const int ntile = wid >> 1;
  const int t0    = blockIdx.x * 64;
  const bool fullT = (t0 + 64 <= T);

  // zero both A fragments once: pad rows (m>=20) stay zero for all k-iters
  {
    int* az = reinterpret_cast<int*>(&afrag[0][0]);  // 512 dwords total
#pragma unroll
    for (int r = 0; r < 2; ++r) az[tid + 256 * r] = 0;
  }

  v8f acc = {0.f, 0.f, 0.f, 0.f, 0.f, 0.f, 0.f, 0.f};

  const int kIters = (Kdim + 31) >> 5;
  for (int kb = 0; kb < kIters; ++kb) {
    const int e0 = kb << 5;

    if (fullT && (e0 + 32 <= Kdim)) {
      // ---------------- fast path: vectorized, guard-free ----------------
      if (tid < 160) {                 // A: 20 rows x 8 chunks
        int m  = tid >> 3;
        int kk = (tid & 7) << 2;
        float4 f = *reinterpret_cast<const float4*>(&C[m * Kdim + e0 + kk]);
        v4h h; h.x = (_Float16)f.x; h.y = (_Float16)f.y;
               h.z = (_Float16)f.z; h.w = (_Float16)f.w;
        int fidx = m >> 4;
        int mloc = m & 15;
        int L    = mloc + (((kk & 15) >= 8) ? 16 : 0);
        int slot = ((kk >= 16) ? 8 : 0) + (kk & 7);
        *reinterpret_cast<v4h*>(&afrag[fidx][L * 16 + slot]) = h;
      }
#pragma unroll
      for (int r = 0; r < 2; ++r) {    // B: 64 rows x 8 chunks
        int q   = tid + 256 * r;
        int row = q >> 3;
        int kk  = (q & 7) << 2;
        int nt  = row >> 4;
        int n   = row & 15;
        int t   = t0 + nt * 16 + n;
        float4 f = *reinterpret_cast<const float4*>(&S[(long)t * Kdim + e0 + kk]);
        v4h h; h.x = (_Float16)f.x; h.y = (_Float16)f.y;
               h.z = (_Float16)f.z; h.w = (_Float16)f.w;
        int L    = n + (((kk & 15) >= 8) ? 16 : 0);
        int slot = ((kk >= 16) ? 8 : 0) + (kk & 7);
        *reinterpret_cast<v4h*>(&bfrag[nt][L * 16 + slot]) = h;
      }
    } else {
      // ---------------- slow path: edge tiles only (rare) ----------------
      if (tid < 160) {
        int m  = tid >> 3;
        int kk0 = (tid & 7) << 2;
#pragma unroll
        for (int u = 0; u < 4; ++u) {
          int kk = kk0 + u;
          int e  = e0 + kk;
          float v = (e < Kdim) ? C[m * Kdim + e] : 0.f;
          int fidx = m >> 4;
          int mloc = m & 15;
          int L    = mloc + (((kk & 15) >= 8) ? 16 : 0);
          int slot = ((kk >= 16) ? 8 : 0) + (kk & 7);
          afrag[fidx][L * 16 + slot] = (_Float16)v;
        }
      }
#pragma unroll
      for (int r = 0; r < 8; ++r) {
        int bi = tid + 256 * r;
        int nt = bi >> 9;
        int rr = bi & 511;
        int n  = rr >> 5;
        int kk = rr & 31;
        int t  = t0 + nt * 16 + n;
        int e  = e0 + kk;
        float v = (t < T && e < Kdim) ? S[(long)t * Kdim + e] : 0.f;
        int L    = n + (((kk & 15) >= 8) ? 16 : 0);
        int slot = ((kk >= 16) ? 8 : 0) + (kk & 7);
        bfrag[nt][L * 16 + slot] = (_Float16)v;
      }
    }
    __syncthreads();

    v16h a = *reinterpret_cast<const v16h*>(&afrag[mtile][lane * 16]);
    v16h b = *reinterpret_cast<const v16h*>(&bfrag[ntile][lane * 16]);
    acc = __builtin_amdgcn_wmma_f32_16x16x32_f16(
        /*neg_a=*/false, a, /*neg_b=*/false, b,
        /*c_mod=*/(short)0, acc, /*reuse_a=*/false, /*reuse_b=*/false);
    __syncthreads();
  }

  // C/D layout: lane L, vgpr v -> M = v + (L>=16?8:0), N = L%16
#pragma unroll
  for (int v = 0; v < 8; ++v) {
    int row = mtile * 16 + v + ((lane >= 16) ? 8 : 0);
    int col = t0 + ntile * 16 + (lane & 15);
    if (row < SUB_NO && col < T) out[(long)row * T + col] = acc[v];
  }
}

// ---------------------------------------------------------------------------
// Kernel 2: rank-12 basis expansion  ke/ki/hist_kern = W(200x12) @ bases(12x200)
// ---------------------------------------------------------------------------
__global__ void basis_expand3(const float* __restrict__ We, const float* __restrict__ Be,
                              const float* __restrict__ Wi, const float* __restrict__ Bi,
                              const float* __restrict__ Wh, const float* __restrict__ Bh,
                              float* __restrict__ ke, float* __restrict__ ki,
                              float* __restrict__ hk) {
  int id = blockIdx.x * blockDim.x + threadIdx.x;
  if (id >= 3 * C_NO * T_NO) return;
  int which = id / (C_NO * T_NO);
  int r = id % (C_NO * T_NO);
  int c = r / T_NO;
  int j = r % T_NO;
  const float* W = (which == 0) ? We : (which == 1) ? Wi : Wh;
  const float* B = (which == 0) ? Be : (which == 1) ? Bi : Bh;
  float acc = 0.f;
#pragma unroll
  for (int b = 0; b < B_NO; ++b) acc = fmaf(W[c * B_NO + b], B[b * T_NO + j], acc);
  float* o = (which == 0) ? ke : (which == 1) ? ki : hk;
  o[r] = acc;
}

// ---------------------------------------------------------------------------
// Kernel 3: causal grouped conv, fused e+i, output S_conv[t, s, m]
// ---------------------------------------------------------------------------
#define CONV_TILE 512
__global__ void grouped_conv(const float* __restrict__ xe_g,  // (20, T)
                             const float* __restrict__ xi_g,  // (20, T)
                             const float* __restrict__ ke,    // (200, 200)
                             const float* __restrict__ ki,    // (200, 200)
                             float* __restrict__ sconv,       // (T, 200)
                             int T) {
  __shared__ float xe[CONV_TILE + 200];
  __shared__ float xi[CONV_TILE + 200];
  __shared__ float kes[M_NO * T_NO];
  __shared__ float kis[M_NO * T_NO];

  const int s   = blockIdx.y;
  const int t0  = blockIdx.x * CONV_TILE;
  const int tid = threadIdx.x;

  for (int j = tid; j < CONV_TILE + 199; j += blockDim.x) {
    int gt = t0 - 199 + j;
    bool ok = (gt >= 0 && gt < T);
    xe[j] = ok ? xe_g[(long)s * T + gt] : 0.f;
    xi[j] = ok ? xi_g[(long)s * T + gt] : 0.f;
  }
  for (int j = tid; j < M_NO * T_NO; j += blockDim.x) {
    kes[j] = ke[(s * M_NO) * T_NO + j];
    kis[j] = ki[(s * M_NO) * T_NO + j];
  }
  __syncthreads();

  for (int idx = tid; idx < M_NO * CONV_TILE; idx += blockDim.x) {
    int m = idx / CONV_TILE;
    int i = idx % CONV_TILE;
    int t = t0 + i;
    if (t >= T) continue;
    const float* kep = &kes[m * T_NO];
    const float* kip = &kis[m * T_NO];
    float acc = 0.f;
#pragma unroll 4
    for (int k = 0; k < T_NO; ++k)
      acc = fmaf(kep[k], xe[i + k], fmaf(kip[k], xi[i + k], acc));
    sconv[(long)t * C_NO + s * M_NO + m] = acc;
  }
}

// ---------------------------------------------------------------------------
// Kernel 4: sequential scan, single workgroup.
//   hist_kern (160 KB) is DMA'd into LDS by the Tensor Data Mover (one wave
//   issues TENSOR_LOAD_TO_LDS, completes with s_wait_tensorcnt 0); history is
//   a double-written ring giving a contiguous 200-tap window. 320 KB WGP LDS.
// ---------------------------------------------------------------------------
#define KSTRIDE 200
#define RSTRIDE 400
__global__ void hist_scan(const float* __restrict__ sconv,  // (T, 200)
                          const float* __restrict__ hk,     // (200, 200)
                          const float* __restrict__ leaf,   // (20, 10)
                          const float* __restrict__ mux,    // (20, 10)
                          const float* __restrict__ bias,   // (20)
                          float* __restrict__ outp, int T) {
  extern __shared__ float sm[];
  float* HK      = sm;                       // 200*200 (exact copy of hk)
  float* ring    = HK + C_NO * KSTRIDE;      // 20*400
  float* part    = ring + SUB_NO * RSTRIDE;  // 400
  float* pr      = part + 400;               // 200
  float* lastout = pr + 200;                 // 32

  const int tid = threadIdx.x;

#if defined(USE_TDM)
  if (tid < 32) {  // wave 0: TDM copy of 200x200 f32 tile, global -> LDS
    unsigned ldsoff = (unsigned)(uintptr_t)(void*)HK;           // LDS byte offset
    unsigned long long ga = (unsigned long long)(uintptr_t)hk;  // global byte addr
    v4u g0;
    g0.x = 1u;                                    // count = 1 valid descriptor
    g0.y = ldsoff;                                // lds_addr
    g0.z = (unsigned)(ga & 0xffffffffu);          // global_addr[31:0]
    g0.w = (unsigned)((ga >> 32) & 0x01ffffffu)   // global_addr[56:32]
           | (2u << 30);                          // type = 2 ("image")
    v8i g1 = { (int)(2u << 16),       // data_size = 2 (4 bytes)
               (int)(200u << 16),     // tensor_dim0[15:0] in bits 79:64.. (=200)
               (int)(200u << 16),     // tensor_dim1[15:0] (=200)
               (int)(200u << 16),     // tile_dim0 = 200
               (int)200,              // tile_dim1 = 200
               (int)200,              // tensor_dim0_stride = 200
               0, 0 };
    v4i gz = {0, 0, 0, 0};
    __builtin_amdgcn_tensor_load_to_lds(g0, g1, gz, gz, 0);
    __builtin_amdgcn_s_wait_tensorcnt(0);
  }
#else
  for (int j = tid; j < C_NO * T_NO; j += blockDim.x) HK[j] = hk[j];
#endif

  for (int j = tid; j < SUB_NO * RSTRIDE; j += blockDim.x) ring[j] = 0.f;
  if (tid < 32) lastout[tid] = 0.f;

  const int half = tid / 200;  // 0 or 1 for tid < 400
  const int c    = tid % 200;
  const int s    = c / M_NO;
  float leafv = 0.f, muxv = 0.f;
  if (tid < 200) { leafv = leaf[c]; muxv = mux[c]; }
  const float biasv = (tid < SUB_NO) ? bias[tid] : 0.f;
  __syncthreads();

  for (int t = 0; t < T; ++t) {
    const int w = t % 200;

    float sc = 0.f;
    if (tid < 200) {
      sc = sconv[(long)t * C_NO + c];  // latency hidden under LDS loop
      __builtin_prefetch(&sconv[(long)(t + 8) * C_NO + c], 0, 0);
    }

    if (tid < 400) {
      const float* Kp = &HK[c * KSTRIDE + half * 100];          // 16B aligned
      const float* Rp = &ring[s * RSTRIDE + w + half * 100];    // contiguous window
      float a0 = 0.f, a1 = 0.f, a2 = 0.f, a3 = 0.f;
      for (int k = 0; k < 100; k += 4) {
        float4 kv = *reinterpret_cast<const float4*>(Kp + k);   // ds_load_b128
        a0 = fmaf(kv.x, Rp[k + 0], a0);
        a1 = fmaf(kv.y, Rp[k + 1], a1);
        a2 = fmaf(kv.z, Rp[k + 2], a2);
        a3 = fmaf(kv.w, Rp[k + 3], a3);
      }
      part[tid] = (a0 + a1) + (a2 + a3);
    }
    __syncthreads();

    if (tid < 200) {
      float hin = part[tid] + part[tid + 200];
      float x   = sc + lastout[s] * leafv + hin;
      pr[c] = tanhf(x) * muxv;
    }
    __syncthreads();

    if (tid < SUB_NO) {
      const float* p = &pr[tid * M_NO];
      float o = biasv;
#pragma unroll
      for (int m = 0; m < M_NO; ++m) o += p[m];
      ring[tid * RSTRIDE + w]       = o;  // double write -> contiguous windows
      ring[tid * RSTRIDE + w + 200] = o;
      lastout[tid] = o;
      if (tid == 0) outp[t] = o;
    }
    __syncthreads();
  }
}

// ---------------------------------------------------------------------------
extern "C" void kernel_launch(void* const* d_in, const int* in_sizes, int n_in,
                              void* d_out, int out_size, void* d_ws, size_t ws_size,
                              hipStream_t stream) {
  const float* S_e      = (const float*)d_in[0];
  const float* S_i      = (const float*)d_in[1];
  // d_in[2] = C_den (unused by reference)
  const float* C_syn_e  = (const float*)d_in[3];
  const float* C_syn_i  = (const float*)d_in[4];
  const float* e_bases  = (const float*)d_in[5];
  const float* e_wts    = (const float*)d_in[6];
  const float* i_bases  = (const float*)d_in[7];
  const float* i_wts    = (const float*)d_in[8];
  const float* h_bases  = (const float*)d_in[9];
  const float* h_wts    = (const float*)d_in[10];
  const float* leaf     = (const float*)d_in[11];
  const float* mux      = (const float*)d_in[12];
  const float* bias     = (const float*)d_in[13];
  float* out = (float*)d_out;

  const int E = in_sizes[3] / SUB_NO;          // 2000
  const int I = in_sizes[4] / SUB_NO;          // 500
  const int T = in_sizes[0] / E;               // 20000

  // workspace carve (floats): ~19.7 MB total
  float* ws     = (float*)d_ws;
  float* Se_sub = ws;                          // 20*T
  float* Si_sub = Se_sub + (size_t)SUB_NO * T;
  float* ke     = Si_sub + (size_t)SUB_NO * T; // 200*200
  float* ki     = ke + C_NO * T_NO;
  float* hk     = ki + C_NO * T_NO;
  float* sconv  = hk + C_NO * T_NO;            // T*200

  // 1) basis expansion (tiny)
  {
    int n = 3 * C_NO * T_NO;
    basis_expand3<<<(n + 255) / 256, 256, 0, stream>>>(
        e_wts, e_bases, i_wts, i_bases, h_wts, h_bases, ke, ki, hk);
  }

  // 2) synapse GEMMs via WMMA (bandwidth-bound, f16 fragments / f32 accum)
  {
    int nblk = (T + 63) / 64;
    syn_gemm_wmma<<<nblk, 256, 0, stream>>>(S_e, C_syn_e, Se_sub, E, T);
    syn_gemm_wmma<<<nblk, 256, 0, stream>>>(S_i, C_syn_i, Si_sub, I, T);
  }

  // 3) causal grouped conv, fused e+i
  {
    dim3 grid((T + CONV_TILE - 1) / CONV_TILE, SUB_NO);
    grouped_conv<<<grid, 256, 0, stream>>>(Se_sub, Si_sub, ke, ki, sconv, T);
  }

  // 4) sequential scan (single workgroup, ~195 KB dynamic LDS — legal on
  //    CDNA5's 320 KB WGP LDS)
  {
    size_t smem = (size_t)(C_NO * KSTRIDE + SUB_NO * RSTRIDE + 400 + 200 + 32)
                  * sizeof(float);
    (void)hipFuncSetAttribute((const void*)hist_scan,
                              hipFuncAttributeMaxDynamicSharedMemorySize,
                              (int)smem);
    hist_scan<<<1, 512, smem, stream>>>(sconv, hk, leaf, mux, bias, out, T);
  }
}
